// TGCNwithAttributes_83202106458536
// MI455X (gfx1250) — matrix operand order repaired and checked by reading, not compiled, not run on hardware
//
#include <hip/hip_runtime.h>

// ---------------------------------------------------------------------------
// Types for WMMA (gfx1250, wave32)
// ---------------------------------------------------------------------------
typedef __attribute__((ext_vector_type(16))) __bf16 v16bf;
typedef __attribute__((ext_vector_type(8)))  float  v8f;

union BFrag {
    uint4 u4[2];   // 32 bytes
    v16bf v;
};

__device__ __forceinline__ unsigned short f32_to_bf16(float x) {
    unsigned int u = __float_as_uint(x);
    u += 0x7FFFu + ((u >> 16) & 1u);      // round-to-nearest-even
    return (unsigned short)(u >> 16);
}

// ---------------------------------------------------------------------------
// K0: deg = 1.0 (self loop), y = 0
// ---------------------------------------------------------------------------
__global__ void k_init(float* __restrict__ deg, float* __restrict__ y,
                       int N, int NY) {
    int i = blockIdx.x * blockDim.x + threadIdx.x;
    if (i < N)  deg[i] = 1.0f;
    if (i < NY) y[i]   = 0.0f;
}

// ---------------------------------------------------------------------------
// K1: x = embed[ids] + pe @ ltW + ltb    (f32 output + bf16 copy for GEMM A)
// one block per node, one thread per feature
// ---------------------------------------------------------------------------
__global__ void k_feat(const int* __restrict__ ids, const float* __restrict__ pe,
                       const float* __restrict__ embed, const float* __restrict__ ltW,
                       const float* __restrict__ ltb,
                       float* __restrict__ x_out, unsigned short* __restrict__ x_bf,
                       int H, int PE) {
    const int i = blockIdx.x;
    const int f = threadIdx.x;
    float acc = embed[(size_t)ids[i] * H + f] + ltb[f];
    for (int p = 0; p < PE; ++p)
        acc += pe[(size_t)i * PE + p] * ltW[(size_t)p * H + f];
    x_out[(size_t)i * H + f] = acc;
    x_bf[(size_t)i * H + f]  = f32_to_bf16(acc);
}

// ---------------------------------------------------------------------------
// K2: weights -> bf16, transposed so B fragments are contiguous in K:
//     W1t[n*H + k] = W1[k*H + n],  W2t[n*H + k] = W2[k*OUT + n]
// ---------------------------------------------------------------------------
__global__ void k_wconv(const float* __restrict__ W1, const float* __restrict__ W2,
                        unsigned short* __restrict__ W1t, unsigned short* __restrict__ W2t,
                        int H, int OUT) {
    const int i  = blockIdx.x * blockDim.x + threadIdx.x;
    const int n1 = H * H;
    if (i < n1) {
        const int n = i / H, k = i % H;
        W1t[i] = f32_to_bf16(W1[(size_t)k * H + n]);
    } else if (i < n1 + OUT * H) {
        const int j = i - n1;
        const int n = j / H, k = j % H;
        W2t[j] = f32_to_bf16(W2[(size_t)k * OUT + n]);
    }
}

// ---------------------------------------------------------------------------
// K3: in-degree accumulation; K4: dinv = rsqrt(deg)
// ---------------------------------------------------------------------------
__global__ void k_deg(const int* __restrict__ dst, float* __restrict__ deg, int E) {
    int e = blockIdx.x * blockDim.x + threadIdx.x;
    if (e < E) atomicAdd(&deg[dst[e]], 1.0f);
}

__global__ void k_dinv(const float* __restrict__ deg, float* __restrict__ dinv, int N) {
    int i = blockIdx.x * blockDim.x + threadIdx.x;
    if (i < N) dinv[i] = rsqrtf(deg[i]);
}

// ---------------------------------------------------------------------------
// K5/K9: C[M,Ncols] = A[M,K](bf16) @ B (Bt[Ncols,K] bf16, pre-transposed)
// One 16x16 output tile per wave; K-loop of V_WMMA_F32_16X16X32_BF16.
// A fragment (ISA 7.12.2, 16-bit A 16x32): lane<16 holds M=lane, K={0..7,16..23};
// lane>=16 holds M=lane-16, K={8..15,24..31}.  B fragment (Bt contiguous in K):
// lane<16 -> N=lane, K={0..7,16..23} ; lane>=16 -> N=lane-16, K={8..15,24..31}.
// D: VGPR r -> row M = r (lanes 0-15) / r+8 (lanes 16-31), N = lane&15.
// ---------------------------------------------------------------------------
__global__ __launch_bounds__(256)
void k_gemm_bf16(const unsigned short* __restrict__ A,
                 const unsigned short* __restrict__ Bt,
                 float* __restrict__ C,
                 int M, int K, int Ncols) {
    const int lane    = threadIdx.x & 31;
    const int wave    = threadIdx.x >> 5;          // 0..7
    const int tiles_n = Ncols >> 4;                // 8 (H=128) or 4 (OUT=64)
    const int mpb     = 8 / tiles_n;               // m-tiles per block
    const int mtile   = blockIdx.x * mpb + wave / tiles_n;
    const int ntile   = wave % tiles_n;
    if (mtile * 16 >= M) return;                   // wave-uniform guard

    int arow = mtile * 16 + (lane & 15);
    if (arow >= M) arow = M - 1;                   // clamp loads only
    const int bcol  = ntile * 16 + (lane & 15);
    const int koffA = (lane < 16) ? 0 : 8;
    const int koffB = (lane < 16) ? 0 : 8;
    const unsigned short* ap = A  + (size_t)arow * K;
    const unsigned short* bp = Bt + (size_t)bcol * K;

    v8f acc = {};
    for (int k0 = 0; k0 < K; k0 += 32) {
        BFrag a, b;
        a.u4[0] = *(const uint4*)(ap + k0 + koffA);
        a.u4[1] = *(const uint4*)(ap + k0 + 16 + koffA);
        b.u4[0] = *(const uint4*)(bp + k0 + koffB);
        b.u4[1] = *(const uint4*)(bp + k0 + 16 + koffB);
        acc = __builtin_amdgcn_wmma_f32_16x16x32_bf16(
                  false, a.v, false, b.v, (short)0, acc, false, false);
    }

    const int drow0 = mtile * 16 + ((lane < 16) ? 0 : 8);
    float* cp = C + (size_t)drow0 * Ncols + ntile * 16 + (lane & 15);
#pragma unroll
    for (int r = 0; r < 8; ++r)
        if (drow0 + r < M) cp[(size_t)r * Ncols] = acc[r];
}

// ---------------------------------------------------------------------------
// K6/K10: agg = h/deg + b   (self-loop term + bias, also initializes agg)
// ---------------------------------------------------------------------------
__global__ void k_selfinit(const float* __restrict__ h, const float* __restrict__ deg,
                           const float* __restrict__ b, float* __restrict__ agg,
                           long long total, int F) {
    long long i = (long long)blockIdx.x * blockDim.x + threadIdx.x;
    if (i >= total) return;
    const int node = (int)(i / F);
    const int f    = (int)(i % F);
    agg[i] = h[i] / deg[node] + b[f];
}

// ---------------------------------------------------------------------------
// K7/K11: edge scatter: agg[dst] += h[src] * dinv[src]*dinv[dst]
// one wave per edge; lanes stride features (coalesced 128B bursts, L2-resident)
// ---------------------------------------------------------------------------
__global__ void k_edge_scatter(const float* __restrict__ h, const float* __restrict__ dinv,
                               const int* __restrict__ src, const int* __restrict__ dst,
                               float* __restrict__ agg, int E, int F) {
    const int lane = threadIdx.x & 31;
    const int e    = blockIdx.x * (blockDim.x >> 5) + (threadIdx.x >> 5);
    if (e >= E) return;
    const int s = src[e], d = dst[e];
    const float w = dinv[s] * dinv[d];
    const float* hrow = h + (size_t)s * F;
    float* arow       = agg + (size_t)d * F;
    for (int f = lane; f < F; f += 32)
        atomicAdd(&arow[f], hrow[f] * w);
}

// ---------------------------------------------------------------------------
// K8: h1r_bf16 = bf16(relu(agg1))
// ---------------------------------------------------------------------------
__global__ void k_relubf(const float* __restrict__ agg, unsigned short* __restrict__ out,
                         long long total) {
    long long i = (long long)blockIdx.x * blockDim.x + threadIdx.x;
    if (i < total) out[i] = f32_to_bf16(fmaxf(agg[i], 0.0f));
}

// ---------------------------------------------------------------------------
// K12: readout y[batch[i]] += agg2[i]
// ---------------------------------------------------------------------------
__global__ void k_readout(const float* __restrict__ agg, const int* __restrict__ batch,
                          float* __restrict__ y, long long total, int F) {
    long long i = (long long)blockIdx.x * blockDim.x + threadIdx.x;
    if (i >= total) return;
    const int node = (int)(i / F);
    const int f    = (int)(i % F);
    atomicAdd(&y[(size_t)batch[node] * F + f], agg[i]);
}

// ---------------------------------------------------------------------------
// Host launcher
// ---------------------------------------------------------------------------
extern "C" void kernel_launch(void* const* d_in, const int* in_sizes, int n_in,
                              void* d_out, int out_size, void* d_ws, size_t ws_size,
                              hipStream_t stream) {
    const int N   = in_sizes[0];            // 100000
    const int E   = in_sizes[2] / 2;        // 1600000
    const int H   = in_sizes[7];            // 128
    const int OUT = in_sizes[11];           // 64
    const int PE  = in_sizes[1] / N;        // 5
    const int NY  = out_size - N * H;       // G*OUT = 65536

    const int*   x_ids = (const int*)d_in[0];
    const float* pe    = (const float*)d_in[1];
    const int*   ei    = (const int*)d_in[2];
    const int*   src   = ei;
    const int*   dst   = ei + E;
    const int*   batch = (const int*)d_in[4];
    const float* embed = (const float*)d_in[5];
    const float* lt_W  = (const float*)d_in[6];
    const float* lt_b  = (const float*)d_in[7];
    const float* W1    = (const float*)d_in[8];
    const float* b1    = (const float*)d_in[9];
    const float* W2    = (const float*)d_in[10];
    const float* b2    = (const float*)d_in[11];

    float* y     = (float*)d_out;                 // [G, OUT]
    float* x_out = y + (size_t)NY;                // [N, H] second tuple output

    // workspace carve (256B aligned), with lifetime-based aliasing
    char*  base = (char*)d_ws;
    size_t off  = 0;
    auto carve = [&](size_t bytes) -> void* {
        void* p = base + off;
        off += (bytes + 255) & ~(size_t)255;
        return p;
    };
    unsigned short* xbf  = (unsigned short*)carve((size_t)N * H * 2); // also h1r bf16
    unsigned short* W1t  = (unsigned short*)carve((size_t)H * H * 2);
    unsigned short* W2t  = (unsigned short*)carve((size_t)OUT * H * 2);
    float* h1   = (float*)carve((size_t)N * H * 4);                   // also h2
    float* agg1 = (float*)carve((size_t)N * H * 4);                   // also agg2
    float* deg  = (float*)carve((size_t)N * 4);
    float* dinv = (float*)carve((size_t)N * 4);
    unsigned short* h1bf = xbf;    // x_bf16 dead after GEMM1
    float* h2   = h1;              // h1 dead after edge scatter 1
    float* agg2 = agg1;            // agg1 dead after relu/bf16

    const int B = 256;
    const long long nh  = (long long)N * H;
    const long long no  = (long long)N * OUT;

    // 0: init deg=1, y=0
    {
        int mx = (N > NY) ? N : NY;
        k_init<<<(mx + B - 1) / B, B, 0, stream>>>(deg, y, N, NY);
    }
    // 1: node features
    k_feat<<<N, H, 0, stream>>>(x_ids, pe, embed, lt_W, lt_b, x_out, xbf, H, PE);
    // 2: weight convert + transpose
    {
        int nW = H * H + OUT * H;
        k_wconv<<<(nW + B - 1) / B, B, 0, stream>>>(W1, W2, W1t, W2t, H, OUT);
    }
    // 3/4: degrees and dinv
    k_deg<<<(E + B - 1) / B, B, 0, stream>>>(dst, deg, E);
    k_dinv<<<(N + B - 1) / B, B, 0, stream>>>(deg, dinv, N);

    // ---- layer 1 ----
    {   // GEMM1: h1 = x @ W1   (tiles_n = H/16, rows per block = 16*(8/tiles_n))
        int tiles_n = H >> 4, rows_pb = 16 * (8 / tiles_n);
        k_gemm_bf16<<<(N + rows_pb - 1) / rows_pb, B, 0, stream>>>(xbf, W1t, h1, N, H, H);
    }
    k_selfinit<<<(int)((nh + B - 1) / B), B, 0, stream>>>(h1, deg, b1, agg1, nh, H);
    k_edge_scatter<<<(E + 7) / 8, B, 0, stream>>>(h1, dinv, src, dst, agg1, E, H);
    k_relubf<<<(int)((nh + B - 1) / B), B, 0, stream>>>(agg1, h1bf, nh);

    // ---- layer 2 ----
    {   // GEMM2: h2 = relu(h1) @ W2
        int tiles_n = OUT >> 4, rows_pb = 16 * (8 / tiles_n);
        k_gemm_bf16<<<(N + rows_pb - 1) / rows_pb, B, 0, stream>>>(h1bf, W2t, h2, N, H, OUT);
    }
    k_selfinit<<<(int)((no + B - 1) / B), B, 0, stream>>>(h2, deg, b2, agg2, no, OUT);
    k_edge_scatter<<<(E + 7) / 8, B, 0, stream>>>(h2, dinv, src, dst, agg2, E, OUT);

    // readout
    k_readout<<<(int)((no + B - 1) / B), B, 0, stream>>>(agg2, batch, y, no, OUT);
}
